// Linear4Bit_86930138071385
// MI455X (gfx1250) — matrix operand order repaired and checked by reading, not compiled, not run on hardware
//
#include <hip/hip_runtime.h>
#include <hip/hip_fp16.h>

// Problem constants (match reference)
constexpr int TOKS  = 8192;
constexpr int INCH  = 4096;
constexpr int OUTCH = 4096;
constexpr int GRP   = 16;             // int4 group size
constexpr int GPB   = GRP / 2;        // 8 packed ints (1 byte each) per group
constexpr int GPROW = INCH / GRP;     // 256 groups per output row

// Tile config: block = 256 threads = 8 waves; C tile = 128(M) x 256(N);
// each wave owns a 64x64 subtile = 4x4 WMMA tiles.
constexpr int BM = 128;
constexpr int BN = 256;

typedef __attribute__((ext_vector_type(8)))  __bf16 v8bf;
typedef __attribute__((ext_vector_type(16))) __bf16 v16bf;
typedef __attribute__((ext_vector_type(8)))  float  v8f;

__launch_bounds__(256)
__global__ void linear4bit_wmma(const float* __restrict__ x,
                                const int* __restrict__ wq4,
                                const __half* __restrict__ wnorm,
                                const float* __restrict__ bias,
                                float* __restrict__ out) {
    constexpr int LDT = 40;  // LDS row stride (ushorts): 32 data + 8 pad
    __shared__ __align__(16) unsigned short As[2][BM * LDT];
    __shared__ __align__(16) unsigned short Bs[2][BN * LDT];

    const int tid   = threadIdx.x;
    const int lane  = tid & 31;
    const int wave  = tid >> 5;
    const int wm    = wave & 1;       // M 64-block within tile (0..1)
    const int wn    = wave >> 1;      // N 64-block within tile (0..3)
    const int l15   = lane & 15;
    const int lhalf = lane >> 4;      // 0: lanes 0-15, 1: lanes 16-31
    const int m0 = blockIdx.y * BM;
    const int n0 = blockIdx.x * BN;

    // staging: thread fills one 16-wide K half of one A row, and the same
    // K half of TWO B output-columns (srow and srow+128)
    const int srow  = tid >> 1;       // 0..127
    const int shalf = tid & 1;        // which 16 K values

    v8f acc[4][4];
    {
        v8f zero = {0.f, 0.f, 0.f, 0.f, 0.f, 0.f, 0.f, 0.f};
#pragma unroll
        for (int mt = 0; mt < 4; ++mt)
#pragma unroll
            for (int nt = 0; nt < 4; ++nt) acc[mt][nt] = zero;
    }

    // global staging registers
    float4 av[4];
    int4   bv[4];
    float  nrm0 = 0.f, nrm1 = 0.f;

    const float*  xrow  = x     + (size_t)(m0 + srow) * INCH + shalf * 16;
    const int*    wrow0 = wq4   + ((size_t)(n0 + srow)       * GPROW + shalf) * GPB;
    const int*    wrow1 = wq4   + ((size_t)(n0 + srow + 128) * GPROW + shalf) * GPB;
    const __half* nrow0 = wnorm + (size_t)(n0 + srow)       * GPROW + shalf;
    const __half* nrow1 = wnorm + (size_t)(n0 + srow + 128) * GPROW + shalf;

    auto load_stage = [&](int kt) {
        const int g = (kt >> 4) * GPB;
        const float4* xp = reinterpret_cast<const float4*>(xrow + kt);
        av[0] = xp[0]; av[1] = xp[1]; av[2] = xp[2]; av[3] = xp[3];
        const int4* wp0 = reinterpret_cast<const int4*>(wrow0 + g);
        const int4* wp1 = reinterpret_cast<const int4*>(wrow1 + g);
        bv[0] = wp0[0]; bv[1] = wp0[1];
        bv[2] = wp1[0]; bv[3] = wp1[1];
        nrm0 = __half2float(nrow0[kt >> 4]);
        nrm1 = __half2float(nrow1[kt >> 4]);
        if (kt + 64 < INCH) {  // prefetch 2 iterations ahead
            __builtin_prefetch(xrow + kt + 64, 0, 0);
            __builtin_prefetch(wrow0 + g + 4 * GPB, 0, 0);
            __builtin_prefetch(wrow1 + g + 4 * GPB, 0, 0);
        }
    };

    // dequant one 16-weight group (8 bytes in 8 ints) into two v8bf
    auto dequant8 = [](const int* bq, float nrm, v8bf& o0, v8bf& o1) {
        const float scale = nrm * (2.0f / 15.0f);
        const float offs  = -nrm;
        v8f lo, hi;
#pragma unroll
        for (int j = 0; j < 4; ++j) {
            const int b = bq[j];
            lo[2 * j]     = (float)(b & 15)  * scale + offs;  // low nibble  -> k=2j
            lo[2 * j + 1] = (float)(b >> 4)  * scale + offs;  // high nibble -> k=2j+1
        }
#pragma unroll
        for (int j = 0; j < 4; ++j) {
            const int b = bq[4 + j];
            hi[2 * j]     = (float)(b & 15)  * scale + offs;
            hi[2 * j + 1] = (float)(b >> 4)  * scale + offs;
        }
        o0 = __builtin_convertvector(lo, v8bf);   // v_cvt_pk_bf16_f32
        o1 = __builtin_convertvector(hi, v8bf);
    };

    auto store_stage = [&](int buf) {
        // ---- A: 16 fp32 -> bf16 via packed hardware converts
        const float* af = reinterpret_cast<const float*>(av);
        v8f a0, a1;
#pragma unroll
        for (int j = 0; j < 8; ++j) { a0[j] = af[j]; a1[j] = af[8 + j]; }
        v8bf* ad = reinterpret_cast<v8bf*>(&As[buf][srow * LDT + shalf * 16]);
        ad[0] = __builtin_convertvector(a0, v8bf);
        ad[1] = __builtin_convertvector(a1, v8bf);
        // ---- B: dequant two groups (cols srow, srow+128)
        v8bf b0, b1;
        dequant8(reinterpret_cast<const int*>(&bv[0]), nrm0, b0, b1);
        v8bf* bd0 = reinterpret_cast<v8bf*>(&Bs[buf][srow * LDT + shalf * 16]);
        bd0[0] = b0; bd0[1] = b1;
        dequant8(reinterpret_cast<const int*>(&bv[2]), nrm1, b0, b1);
        v8bf* bd1 = reinterpret_cast<v8bf*>(&Bs[buf][(srow + 128) * LDT + shalf * 16]);
        bd1[0] = b0; bd1[1] = b1;
    };

    load_stage(0);
    int cur = 0;
    constexpr int NITER = INCH / 32;
    for (int it = 0; it < NITER; ++it) {
        store_stage(cur);
        __syncthreads();                          // single barrier per iteration
        if (it + 1 < NITER) load_stage((it + 1) * 32);

        const unsigned short* Ab = As[cur];
        const unsigned short* Bb = Bs[cur];

        // A fragments: 16-bit A 16x32 layout (ISA 7.12.2)
        v16bf afr[4];
#pragma unroll
        for (int mt = 0; mt < 4; ++mt) {
            const int r = wm * 64 + mt * 16 + l15;
            const unsigned short* pa = &Ab[r * LDT + lhalf * 8];
            v8bf lo = *reinterpret_cast<const v8bf*>(pa);        // K 0-7   / 8-15
            v8bf hi = *reinterpret_cast<const v8bf*>(pa + 16);   // K 16-23 / 24-31
            afr[mt] = __builtin_shufflevector(lo, hi, 0,1,2,3,4,5,6,7,8,9,10,11,12,13,14,15);
        }
        // B fragments: lanes 0-15 K 0-7,16-23 at N=lane; lanes 16-31 K 8-15,24-31
        v16bf bfr[4];
#pragma unroll
        for (int nt = 0; nt < 4; ++nt) {
            const int c = wn * 64 + nt * 16 + l15;
            const unsigned short* pb = &Bb[c * LDT + lhalf * 8];
            v8bf lo = *reinterpret_cast<const v8bf*>(pb);
            v8bf hi = *reinterpret_cast<const v8bf*>(pb + 16);
            bfr[nt] = __builtin_shufflevector(lo, hi, 0,1,2,3,4,5,6,7,8,9,10,11,12,13,14,15);
        }

#pragma unroll
        for (int mt = 0; mt < 4; ++mt)
#pragma unroll
            for (int nt = 0; nt < 4; ++nt)
                acc[mt][nt] = __builtin_amdgcn_wmma_f32_16x16x32_bf16(
                    false, afr[mt], false, bfr[nt], (short)0, acc[mt][nt], false, false);

        cur ^= 1;
    }

    // Epilogue: C/D layout — VGPR r: lanes 0-15 M=r, lanes 16-31 M=8+r; N = lane&15
#pragma unroll
    for (int nt = 0; nt < 4; ++nt) {
        const int n = n0 + wn * 64 + nt * 16 + l15;
        const float bvs = bias[n];
#pragma unroll
        for (int mt = 0; mt < 4; ++mt) {
            const int rb = m0 + wm * 64 + mt * 16 + lhalf * 8;
#pragma unroll
            for (int r = 0; r < 8; ++r)
                out[(size_t)(rb + r) * OUTCH + n] = acc[mt][nt][r] + bvs;
        }
    }
}

extern "C" void kernel_launch(void* const* d_in, const int* in_sizes, int n_in,
                              void* d_out, int out_size, void* d_ws, size_t ws_size,
                              hipStream_t stream) {
    (void)in_sizes; (void)n_in; (void)out_size; (void)d_ws; (void)ws_size;
    const float*  x    = (const float*)d_in[0];
    const int*    wq4  = (const int*)d_in[1];
    const __half* wnrm = (const __half*)d_in[2];
    const float*  bias = (const float*)d_in[3];
    float* out = (float*)d_out;

    dim3 grid(OUTCH / BN, TOKS / BM);   // 16 x 64 = 1024 blocks
    linear4bit_wmma<<<grid, 256, 0, stream>>>(x, wq4, wnrm, bias, out);
}
